// VanillaRNN_50938312130618
// MI455X (gfx1250) — matrix operand order repaired and checked by reading, not compiled, not run on hardware
//
#include <hip/hip_runtime.h>

// ---------------------------------------------------------------------------
// Vanilla RNN on MI455X (gfx1250, wave32, WMMA).
//   h_{t+1} = tanh(x_t (outer) W_hx + h_t @ W_hh + b_h), t = 0..63
//   out     = h_T @ W_ph^T + b_p
// v3: 2x2 register blocking per wave (4 WMMAs per 8 ds_load_b128), 128-thread
//     WGs keep 256 WGs/step. W_hh pre-transposed to f16 [N][K]; BK=64;
//     double-buffered LDS with a single barrier per chunk.
// ---------------------------------------------------------------------------

typedef __attribute__((ext_vector_type(16))) _Float16 v16h;
typedef __attribute__((ext_vector_type(8)))  float    v8f;

#define RNN_B 512
#define RNN_T 64
#define RNN_H 2048
#define RNN_C 10

#define BM 64
#define BN 64
#define BK 64
#define LDK 72                 // BK + 8 halfword pad; multiple of 8 keeps 16B align
#define NCHUNK (RNN_H / BK)    // 32

// POD 16-byte vector (avoid HIP uint4 ctor issues inside unions)
struct alignas(16) U128 { unsigned int x, y, z, w; };

union Frag16 {
    U128 q[2];
    v16h h;
};

__device__ __forceinline__ unsigned short f32_to_f16u(float f) {
    _Float16 h = (_Float16)f;
    return __builtin_bit_cast(unsigned short, h);
}
__device__ __forceinline__ float f16u_to_f32(unsigned short u) {
    return (float)__builtin_bit_cast(_Float16, u);
}

// ---- one-time: W_hh fp32 [K][N] -> f16 transposed [N][K] -------------------
__global__ __launch_bounds__(256)
void cvt_transpose_whh(const float* __restrict__ in,        // [H][H], row = k
                       unsigned short* __restrict__ out) {  // [H][H], row = n
    __shared__ unsigned short tile[32][33];
    const int nb = blockIdx.x * 32;
    const int kb = blockIdx.y * 32;
    const int tx = threadIdx.x & 31;
    const int ty = threadIdx.x >> 5;          // 0..7
    #pragma unroll
    for (int i = 0; i < 32; i += 8)
        tile[ty + i][tx] = f32_to_f16u(in[(size_t)(kb + ty + i) * RNN_H + nb + tx]);
    __syncthreads();
    #pragma unroll
    for (int i = 0; i < 32; i += 8)
        out[(size_t)(nb + ty + i) * RNN_H + kb + tx] = tile[tx][ty + i];
}

// ---- one-time: h0 = 0 ------------------------------------------------------
__global__ void zero_f16(unsigned short* __restrict__ out, int n) {
    int i = blockIdx.x * blockDim.x + threadIdx.x;
    if (i < n) out[i] = 0;
}

// ---- per-timestep fused GEMM + rank-1 + tanh -------------------------------
// h_out[b][j] = tanh( x[b][t]*W_hx[j] + b_h[j] + sum_k h_in[b][k]*W_hh[k][j] )
// Grid: (H/BN, B/BM) = 32x8 = 256 WGs, 128 threads (4 waves).
// Each wave: 2x2 grid of 16x16 C tiles (32x32 quadrant of the 64x64 WG tile).
__global__ __launch_bounds__(128)
void rnn_step(const unsigned short* __restrict__ hin,   // [B][H] f16
              unsigned short* __restrict__ hout,        // [B][H] f16
              const unsigned short* __restrict__ whhT,  // [H][H] f16, row = n
              const float* __restrict__ x,              // [B][T] f32
              const float* __restrict__ whx,            // [H]    f32
              const float* __restrict__ bh,             // [H]    f32
              int t)
{
    __shared__ __align__(16) unsigned short As[2][BM * LDK]; // [m][k]
    __shared__ __align__(16) unsigned short Bs[2][BN * LDK]; // [n][k]

    const int tid  = threadIdx.x;
    const int lane = tid & 31;
    const int wid  = tid >> 5;        // 0..3
    const int wr   = wid >> 1;        // 0..1  -> M offset 32*wr
    const int wc   = wid & 1;         // 0..1  -> N offset 32*wc
    const int r    = lane & 15;
    const int hsel = lane >> 4;       // lane-half select

    const int rowBlk = blockIdx.y * BM;   // batch offset
    const int colBlk = blockIdx.x * BN;   // hidden-unit offset

    // Tile fill: 64 rows x 64 halfwords; per thread 4x b128 (A and B identical)
    const int f_row = tid >> 1;           // 0..63
    const int f_cg  = (tid & 1) * 32;     // halfword col offset 0/32
    const int fillOff = f_row * LDK + f_cg;

    const unsigned short* aSrc = hin  + (size_t)(rowBlk + f_row) * RNN_H + f_cg;
    const unsigned short* bSrc = whhT + (size_t)(colBlk + f_row) * RNN_H + f_cg;

    // prologue: stage chunk 0 in registers
    U128 a0 = *(const U128*)(aSrc);
    U128 a1 = *(const U128*)(aSrc + 8);
    U128 a2 = *(const U128*)(aSrc + 16);
    U128 a3 = *(const U128*)(aSrc + 24);
    U128 b0 = *(const U128*)(bSrc);
    U128 b1 = *(const U128*)(bSrc + 8);
    U128 b2 = *(const U128*)(bSrc + 16);
    U128 b3 = *(const U128*)(bSrc + 24);

    v8f acc00 = {}, acc01 = {}, acc10 = {}, acc11 = {};

    for (int c = 0; c < NCHUNK; ++c) {
        unsigned short* A  = As[c & 1];
        unsigned short* Bt = Bs[c & 1];
        *(U128*)(A  + fillOff)      = a0;
        *(U128*)(A  + fillOff + 8)  = a1;
        *(U128*)(A  + fillOff + 16) = a2;
        *(U128*)(A  + fillOff + 24) = a3;
        *(U128*)(Bt + fillOff)      = b0;
        *(U128*)(Bt + fillOff + 8)  = b1;
        *(U128*)(Bt + fillOff + 16) = b2;
        *(U128*)(Bt + fillOff + 24) = b3;

        if (c + 1 < NCHUNK) {
            const unsigned short* an = aSrc + (size_t)(c + 1) * BK;
            const unsigned short* bn = bSrc + (size_t)(c + 1) * BK;
            a0 = *(const U128*)(an);
            a1 = *(const U128*)(an + 8);
            a2 = *(const U128*)(an + 16);
            a3 = *(const U128*)(an + 24);
            b0 = *(const U128*)(bn);
            b1 = *(const U128*)(bn + 8);
            b2 = *(const U128*)(bn + 16);
            b3 = *(const U128*)(bn + 24);
            if (c + 2 < NCHUNK)
                __builtin_prefetch(bn + BK, 0, 1);   // global_prefetch_b8
        }
        __syncthreads();   // chunk c visible; prior reads of this buffer done

        #pragma unroll
        for (int s = 0; s < 2; ++s) {   // two 32-wide K steps per chunk
            // A 16x32: lanes 0-15 K in {0-7,16-23}, lanes 16-31 {8-15,24-31}
            Frag16 fa0, fa1, fb0, fb1;
            const unsigned short* ap0 = A + (wr * 32 + r) * LDK + s * 32 + 8 * hsel;
            fa0.q[0] = *(const U128*)(ap0);
            fa0.q[1] = *(const U128*)(ap0 + 16);
            const unsigned short* ap1 = ap0 + 16 * LDK;
            fa1.q[0] = *(const U128*)(ap1);
            fa1.q[1] = *(const U128*)(ap1 + 16);
            // B 32x16: lanes 0-15 K in {0-7,16-23}, lanes 16-31 {8-15,24-31}
            const unsigned short* bp0 = Bt + (wc * 32 + r) * LDK + s * 32 + 8 * hsel;
            fb0.q[0] = *(const U128*)(bp0);
            fb0.q[1] = *(const U128*)(bp0 + 16);
            const unsigned short* bp1 = bp0 + 16 * LDK;
            fb1.q[0] = *(const U128*)(bp1);
            fb1.q[1] = *(const U128*)(bp1 + 16);

            acc00 = __builtin_amdgcn_wmma_f32_16x16x32_f16(
                        false, fa0.h, false, fb0.h, (short)0, acc00, false, false);
            acc01 = __builtin_amdgcn_wmma_f32_16x16x32_f16(
                        false, fa0.h, false, fb1.h, (short)0, acc01, false, false);
            acc10 = __builtin_amdgcn_wmma_f32_16x16x32_f16(
                        false, fa1.h, false, fb0.h, (short)0, acc10, false, false);
            acc11 = __builtin_amdgcn_wmma_f32_16x16x32_f16(
                        false, fa1.h, false, fb1.h, (short)0, acc11, false, false);
        }
    }

    // ---- epilogue: + x_t*W_hx + b_h, tanh, f16 store ----
    // C layout: VGPR i -> M = i + 8*hsel, N = lane&15
    const int n0 = colBlk + wc * 32 + r;
    const int n1 = n0 + 16;
    const float w0  = whx[n0];
    const float w1  = whx[n1];
    const float bh0 = bh[n0];
    const float bh1 = bh[n1];

    #pragma unroll
    for (int ai = 0; ai < 2; ++ai) {
        const int m_base = rowBlk + wr * 32 + ai * 16 + 8 * hsel;
        const v8f accL = ai ? acc10 : acc00;
        const v8f accR = ai ? acc11 : acc01;
        #pragma unroll
        for (int i = 0; i < 8; ++i) {
            const int b  = m_base + i;
            const float xv = x[(size_t)b * RNN_T + t];
            const float v0 = tanhf(accL[i] + xv * w0 + bh0);
            const float v1 = tanhf(accR[i] + xv * w1 + bh1);
            hout[(size_t)b * RNN_H + n0] = f32_to_f16u(v0);
            hout[(size_t)b * RNN_H + n1] = f32_to_f16u(v1);
        }
    }
}

// ---- output projection: out[b][c] = b_p[c] + sum_j h[b][j]*W_ph[c][j] -----
__global__ __launch_bounds__(256)
void proj(const unsigned short* __restrict__ h,   // [B][H] f16
          const float* __restrict__ wph,          // [C][H]
          const float* __restrict__ bp,           // [C]
          float* __restrict__ out)                // [B][C]
{
    const int b   = blockIdx.x;
    const int tid = threadIdx.x;

    float part[RNN_C];
    #pragma unroll
    for (int c = 0; c < RNN_C; ++c) part[c] = 0.0f;

    for (int j = tid; j < RNN_H; j += 256) {
        const float hv = f16u_to_f32(h[(size_t)b * RNN_H + j]);
        #pragma unroll
        for (int c = 0; c < RNN_C; ++c) part[c] += hv * wph[(size_t)c * RNN_H + j];
    }

    #pragma unroll
    for (int c = 0; c < RNN_C; ++c)
        for (int off = 16; off > 0; off >>= 1)
            part[c] += __shfl_down(part[c], off, 32);

    __shared__ float red[8][RNN_C];
    const int lane = tid & 31, wv = tid >> 5;
    if (lane == 0) {
        #pragma unroll
        for (int c = 0; c < RNN_C; ++c) red[wv][c] = part[c];
    }
    __syncthreads();
    if (tid < RNN_C) {
        float s = bp[tid];
        #pragma unroll
        for (int w = 0; w < 8; ++w) s += red[w][tid];
        out[(size_t)b * RNN_C + tid] = s;
    }
}

// ---------------------------------------------------------------------------
extern "C" void kernel_launch(void* const* d_in, const int* in_sizes, int n_in,
                              void* d_out, int out_size, void* d_ws, size_t ws_size,
                              hipStream_t stream) {
    const float* x   = (const float*)d_in[0];   // [B][T]
    const float* whx = (const float*)d_in[1];   // [H][1]
    const float* whh = (const float*)d_in[2];   // [H][H]
    const float* bh  = (const float*)d_in[3];   // [H]
    const float* wph = (const float*)d_in[4];   // [C][H]
    const float* bp  = (const float*)d_in[5];   // [C]
    float* out = (float*)d_out;

    // workspace: W_hh^T f16 (8 MB) | hA f16 (2 MB) | hB f16 (2 MB)
    unsigned short* whhT = (unsigned short*)d_ws;
    unsigned short* hA   = whhT + (size_t)RNN_H * RNN_H;
    unsigned short* hB   = hA + (size_t)RNN_B * RNN_H;

    {
        dim3 g(RNN_H / 32, RNN_H / 32);
        cvt_transpose_whh<<<g, 256, 0, stream>>>(whh, whhT);
    }
    {
        int n = RNN_B * RNN_H;
        zero_f16<<<(n + 255) / 256, 256, 0, stream>>>(hA, n);
    }

    unsigned short* cur = hA;
    unsigned short* nxt = hB;
    dim3 grid(RNN_H / BN, RNN_B / BM);   // 32 x 8 = 256 workgroups/step
    for (int t = 0; t < RNN_T; ++t) {
        rnn_step<<<grid, 128, 0, stream>>>(cur, nxt, whhT, x, whx, bh, t);
        unsigned short* tmp = cur; cur = nxt; nxt = tmp;
    }

    proj<<<RNN_B, 256, 0, stream>>>(cur, wph, bp, out);
}
